// MultiHeadAttentionBlock_55173149884718
// MI455X (gfx1250) — matrix operand order repaired and checked by reading, not compiled, not run on hardware
//
#include <hip/hip_runtime.h>

// ---------------------------------------------------------------------------
// MI455X (gfx1250) MHA block: QKV proj -> flash attention -> O proj
// Matmuls via v_wmma_f32_16x16x32_bf16 (fp32 accumulate).
// Attention K/V tiles streamed with global_load_async_to_lds_b128 (ASYNCcnt)
// into double-buffered LDS, overlapping HBM traffic with WMMA compute.
// ---------------------------------------------------------------------------

typedef __attribute__((ext_vector_type(8)))  float  v8f;
typedef __attribute__((ext_vector_type(16))) __bf16 v16bf;

union Frag16 { v16bf v; unsigned u[8]; };

#define D_MODEL 1024
#define N_HEADS 16
#define DK      64
#define BATCH   2
#define SEQ     2048
#define ROWS    (BATCH * SEQ)   // 4096
#define NTILES  (SEQ / 128)     // 16 key tiles

// ----- bf16 helpers (round-to-nearest-even) --------------------------------
__device__ __forceinline__ unsigned short f2bf(float f) {
    unsigned u = __builtin_bit_cast(unsigned, f);
    u += 0x7FFFu + ((u >> 16) & 1u);
    return (unsigned short)(u >> 16);
}
__device__ __forceinline__ unsigned f2bf2(float a, float b) {
    unsigned ua = __builtin_bit_cast(unsigned, a);
    unsigned ub = __builtin_bit_cast(unsigned, b);
    ua += 0x7FFFu + ((ua >> 16) & 1u);
    ub += 0x7FFFu + ((ub >> 16) & 1u);
    return (ua >> 16) | (ub & 0xFFFF0000u);
}
__device__ __forceinline__ float cvtF(float x) { return x; }
__device__ __forceinline__ float cvtF(unsigned short x) {
    unsigned u = (unsigned)x << 16;
    return __builtin_bit_cast(float, u);
}

// Fragment u32-index helpers per CDNA5 ISA 7.12.2 VGPR layouts (16-bit data):
//   A 16x32: lane m=lane&15; VGPR j holds K = 2j + (j>=4?8:0) + (lane>>4)*8 .. +1
//   B 32x16: lane n=lane&15; VGPR j holds K = 2j + (j>=4?8:0) + (lane>>4)*8 .. +1
__device__ __forceinline__ int aidx(int j, int half) { return j + ((j >= 4) ? 4 : 0) + half * 4; }
__device__ __forceinline__ int bidx(int j, int half) { return j + ((j >= 4) ? 4 : 0) + half * 4; }

// ----- CDNA5 async global->LDS copy (VGLOBAL op 98, GVS mode, ASYNCcnt) ----
__device__ __forceinline__ void async_copy_b128(unsigned lds_byte_addr,
                                                unsigned g_byte_off,
                                                const void* sbase) {
    asm volatile("global_load_async_to_lds_b128 %0, %1, %2"
                 :: "v"(lds_byte_addr), "v"(g_byte_off), "s"(sbase)
                 : "memory");
}
__device__ __forceinline__ void wait_async_all() {
    asm volatile("s_wait_asynccnt 0" ::: "memory");
}

// ---------------------------------------------------------------------------
// Tiled GEMM:  Out = A(MxK) @ W(KxN) + bias,  bf16 WMMA, fp32 accum.
// MODE 0: Out = bf16 at [b,h,l,d]   (Q and K projections)
// MODE 2: Out = bf16 at [b,h,d,l]   (V projection, transposed)
// MODE 3: Out = fp32 at [m,n]       (final O projection)
// Block: 256 threads (8 waves); tile 128x128, K-step 32. A/W are converted
// fp32->bf16 through VGPRs (conversion forbids pure async copy here).
// ---------------------------------------------------------------------------
template <typename AT, int MODE>
__global__ __launch_bounds__(256)
void gemm_bf16_wmma(const AT* __restrict__ A, const float* __restrict__ W,
                    const float* __restrict__ bias, void* __restrict__ Out,
                    int M, int K, int N) {
    __shared__ unsigned As[128 * 16];   // [m][k-pair], 8 KB
    __shared__ unsigned Bs[128 * 16];   // [n][k-pair], 8 KB

    const int tid  = threadIdx.x;
    const int lane = tid & 31;
    const int wave = tid >> 5;
    const int m_l  = lane & 15;
    const int half = lane >> 4;
    const int mtile = blockIdx.y * 128;
    const int ntile = blockIdx.x * 128;

    v8f acc[8] = {};

    for (int kb = 0; kb < K; kb += 32) {
        __syncthreads();
        {   // A tile -> LDS (convert to bf16, pack pairs along K)
            const int m  = tid >> 1;
            const int k0 = (tid & 1) * 16;
            const AT* src = A + (size_t)(mtile + m) * K + kb + k0;
            unsigned* dst = &As[m * 16 + (k0 >> 1)];
#pragma unroll
            for (int i = 0; i < 8; ++i)
                dst[i] = f2bf2(cvtF(src[2 * i]), cvtF(src[2 * i + 1]));
            if (kb + 32 < K) __builtin_prefetch(src + 32, 0, 0);
        }
        {   // W tile -> LDS transposed to [n][k] so K-pairs are contiguous
            const int n   = tid & 127;
            const int kk0 = (tid >> 7) * 16;
            const float* src = W + (size_t)(kb + kk0) * N + ntile + n;
            unsigned* dst = &Bs[n * 16 + (kk0 >> 1)];
#pragma unroll
            for (int i = 0; i < 8; ++i)
                dst[i] = f2bf2(src[(size_t)(2 * i) * N], src[(size_t)(2 * i + 1) * N]);
            if (kb + 32 < K) __builtin_prefetch(src + (size_t)32 * N, 0, 0);
        }
        __syncthreads();

        Frag16 af;
#pragma unroll
        for (int j = 0; j < 8; ++j)
            af.u[j] = As[(wave * 16 + m_l) * 16 + aidx(j, half)];
#pragma unroll
        for (int f = 0; f < 8; ++f) {
            Frag16 bf_;
#pragma unroll
            for (int j = 0; j < 8; ++j)
                bf_.u[j] = Bs[(f * 16 + m_l) * 16 + bidx(j, half)];
            acc[f] = __builtin_amdgcn_wmma_f32_16x16x32_bf16(
                false, af.v, false, bf_.v, (short)0, acc[f], false, false);
        }
    }

    // Epilogue: bias + store. C/D layout: lane holds col n = f*16 + m_l,
    // VGPR r holds row mtile + wave*16 + half*8 + r.
#pragma unroll
    for (int f = 0; f < 8; ++f) {
        const int ncol = ntile + f * 16 + m_l;
        const float bv = bias[ncol];
#pragma unroll
        for (int r = 0; r < 8; ++r) {
            const int mrow = mtile + wave * 16 + half * 8 + r;
            const float val = acc[f][r] + bv;
            if (MODE == 3) {
                ((float*)Out)[(size_t)mrow * N + ncol] = val;
            } else {
                const int b = mrow >> 11, l = mrow & (SEQ - 1);
                const int h = ncol >> 6,  d = ncol & (DK - 1);
                size_t idx;
                if (MODE == 2) idx = (((size_t)(b * N_HEADS + h) * DK + d) * SEQ + l);
                else           idx = (((size_t)(b * N_HEADS + h) * SEQ + l) * DK + d);
                ((unsigned short*)Out)[idx] = f2bf(val);
            }
        }
    }
}

// ---------------------------------------------------------------------------
// Flash attention. Grid: (SEQ/128, B*H). Block 256 = 8 waves; each wave owns
// 16 query rows. Streams 128-key tiles with online softmax; scores never
// touch HBM. Q,K are [l,d] bf16 per head; V is [d,l] bf16 per head.
// K/V tiles double-buffered in LDS via async-to-LDS; tile t+1 streams while
// tile t is consumed by WMMA. Output O written bf16 at [b,l,D_MODEL].
// ---------------------------------------------------------------------------
__global__ __launch_bounds__(256)
void flash_attn_wmma(const unsigned short* __restrict__ Qb,
                     const unsigned short* __restrict__ Kb,
                     const unsigned short* __restrict__ Vt,
                     unsigned short* __restrict__ O) {
    __shared__ unsigned Ks[2][128 * 32]; // [n][d-pair]  2 x 16 KB
    __shared__ unsigned Vs[2][64 * 64];  // [d][n-pair]  2 x 16 KB
    __shared__ unsigned Ps[8][16 * 64];  // per-wave P tile, [m][n-pair] 32 KB

    const int tid  = threadIdx.x;
    const int lane = tid & 31;
    const int wave = tid >> 5;
    const int m_l  = lane & 15;
    const int half = lane >> 4;
    const int bh    = blockIdx.y;           // b*16 + h
    const int qtile = blockIdx.x * 128;
    const size_t hoff = (size_t)bh * SEQ * DK;
    const unsigned short* Qh = Qb + hoff;
    const unsigned short* Kh = Kb + hoff;
    const unsigned short* Vh = Vt + hoff;

    // Issue one 128-key tile (K: contiguous 16 KB; V: 64 rows x 256 B) as
    // async global->LDS b128 copies: 8 instructions per thread, ASYNCcnt.
    auto issue_tile = [&](int n0, int buf) {
        const void* kbase = (const void*)(Kh + (size_t)n0 * DK);
        const unsigned ks0 = (unsigned)(size_t)(void*)&Ks[buf][0];
#pragma unroll
        for (int i = 0; i < 4; ++i) {
            const unsigned off = (unsigned)(i * 256 + tid) * 16u;
            async_copy_b128(ks0 + off, off, kbase);
        }
        const void* vbase = (const void*)(Vh + (size_t)n0);
        const unsigned vs0 = (unsigned)(size_t)(void*)&Vs[buf][0];
#pragma unroll
        for (int i = 0; i < 4; ++i) {
            const int c = i * 256 + tid;          // 0..1023 16B chunks
            const int d = c >> 4, w16 = c & 15;
            const unsigned goff = (unsigned)(d * (SEQ * 2) + w16 * 16);
            async_copy_b128(vs0 + (unsigned)c * 16u, goff, vbase);
        }
    };

    // Q fragments: two 16x32 A-frags covering d = 0..63, loaded once.
    Frag16 qf[2];
    {
        const int qrow = qtile + wave * 16 + m_l;
        const unsigned* src = (const unsigned*)(Qh + (size_t)qrow * DK);
#pragma unroll
        for (int kk = 0; kk < 2; ++kk)
#pragma unroll
            for (int j = 0; j < 8; ++j)
                qf[kk].u[j] = src[kk * 16 + aidx(j, half)];
    }

    float mrow[8], lrow[8];
    v8f acc[4] = {};
#pragma unroll
    for (int r = 0; r < 8; ++r) { mrow[r] = -3.0e38f; lrow[r] = 0.0f; }

    issue_tile(0, 0);                       // prologue: stage tile 0

    for (int t = 0; t < NTILES; ++t) {
        const int buf = t & 1;
        wait_async_all();                   // my tile-t copies complete
        __syncthreads();                    // tile t visible to all waves;
                                            // everyone past compute t-1
        if (t + 1 < NTILES)                 // stream tile t+1 during compute t
            issue_tile((t + 1) * 128, buf ^ 1);

        const unsigned* Kst = &Ks[buf][0];
        const unsigned* Vst = &Vs[buf][0];

        // S = (Q K^T) * 1/sqrt(dk)
        v8f s[8];
#pragma unroll
        for (int f = 0; f < 8; ++f) {
            v8f c = {};
#pragma unroll
            for (int kk = 0; kk < 2; ++kk) {
                Frag16 kfr;
#pragma unroll
                for (int j = 0; j < 8; ++j)
                    kfr.u[j] = Kst[(f * 16 + m_l) * 32 + kk * 16 + bidx(j, half)];
                c = __builtin_amdgcn_wmma_f32_16x16x32_bf16(
                    false, qf[kk].v, false, kfr.v, (short)0, c, false, false);
            }
#pragma unroll
            for (int r = 0; r < 8; ++r) c[r] *= 0.125f;
            s[f] = c;
        }

        // Online softmax: row max across 8 frags then across the 16-lane half.
        float mnew[8];
#pragma unroll
        for (int r = 0; r < 8; ++r) {
            float v = s[0][r];
#pragma unroll
            for (int f = 1; f < 8; ++f) v = fmaxf(v, s[f][r]);
#pragma unroll
            for (int xm = 1; xm < 16; xm <<= 1)
                v = fmaxf(v, __shfl_xor(v, xm, 32));
            mnew[r] = fmaxf(v, mrow[r]);
        }

        float rs[8];
#pragma unroll
        for (int r = 0; r < 8; ++r) {
            const float sc = __expf(mrow[r] - mnew[r]);
            mrow[r] = mnew[r];
            lrow[r] *= sc;
            rs[r] = 0.0f;
#pragma unroll
            for (int fd = 0; fd < 4; ++fd) acc[fd][r] *= sc;
        }

        // P = exp(S - m); stash bf16 P in per-wave LDS in A-fragment layout.
        unsigned* Pw = &Ps[wave][0];
#pragma unroll
        for (int f = 0; f < 8; ++f) {
#pragma unroll
            for (int r = 0; r < 8; ++r) {
                const float p = __expf(s[f][r] - mrow[r]);
                rs[r] += p;
                ((unsigned short*)Pw)[(half * 8 + r) * 128 + f * 16 + m_l] = f2bf(p);
            }
        }
#pragma unroll
        for (int r = 0; r < 8; ++r) {
            float v = rs[r];
#pragma unroll
            for (int xm = 1; xm < 16; xm <<= 1) v += __shfl_xor(v, xm, 32);
            lrow[r] += v;
        }

        // acc += P @ V  (inner dim = 128 keys, 4 K-steps of 32)
#pragma unroll
        for (int fd = 0; fd < 4; ++fd) {
#pragma unroll
            for (int kb2 = 0; kb2 < 4; ++kb2) {
                Frag16 pa, vb;
#pragma unroll
                for (int j = 0; j < 8; ++j) {
                    pa.u[j] = Pw[m_l * 64 + kb2 * 16 + aidx(j, half)];
                    vb.u[j] = Vst[(fd * 16 + m_l) * 64 + kb2 * 16 + bidx(j, half)];
                }
                acc[fd] = __builtin_amdgcn_wmma_f32_16x16x32_bf16(
                    false, pa.v, false, vb.v, (short)0, acc[fd], false, false);
            }
        }
    }

    // Normalize and write O as bf16 at [b, l, h*64 + d].
    const int b = bh >> 4, h = bh & 15;
#pragma unroll
    for (int fd = 0; fd < 4; ++fd) {
        const int d = fd * 16 + m_l;
#pragma unroll
        for (int r = 0; r < 8; ++r) {
            const int l = qtile + wave * 16 + half * 8 + r;
            const float val = acc[fd][r] / lrow[r];
            O[(size_t)(b * SEQ + l) * D_MODEL + h * DK + d] = f2bf(val);
        }
    }
}

// ---------------------------------------------------------------------------
extern "C" void kernel_launch(void* const* d_in, const int* in_sizes, int n_in,
                              void* d_out, int out_size, void* d_ws, size_t ws_size,
                              hipStream_t stream) {
    (void)in_sizes; (void)n_in; (void)out_size; (void)ws_size;

    const float* q   = (const float*)d_in[0];
    const float* k   = (const float*)d_in[1];
    const float* v   = (const float*)d_in[2];
    const float* w_q = (const float*)d_in[3];
    const float* b_q = (const float*)d_in[4];
    const float* w_k = (const float*)d_in[5];
    const float* b_k = (const float*)d_in[6];
    const float* w_v = (const float*)d_in[7];
    const float* b_v = (const float*)d_in[8];
    const float* w_o = (const float*)d_in[9];
    const float* b_o = (const float*)d_in[10];

    // Workspace layout (bf16): Qb[b,h,l,d] | Kb[b,h,l,d] | Vt[b,h,d,l] | O[b,l,D]
    const size_t SLAB = (size_t)BATCH * N_HEADS * SEQ * DK * sizeof(unsigned short); // 8 MB
    char* ws = (char*)d_ws;
    unsigned short* Qb = (unsigned short*)(ws);
    unsigned short* Kb = (unsigned short*)(ws + SLAB);
    unsigned short* Vt = (unsigned short*)(ws + 2 * SLAB);
    unsigned short* O  = (unsigned short*)(ws + 3 * SLAB);

    const dim3 gemm_grid(D_MODEL / 128, ROWS / 128);   // (8, 32)
    const dim3 attn_grid(SEQ / 128, BATCH * N_HEADS);  // (16, 32)

    gemm_bf16_wmma<float, 0><<<gemm_grid, 256, 0, stream>>>(
        q, w_q, b_q, (void*)Qb, ROWS, D_MODEL, D_MODEL);
    gemm_bf16_wmma<float, 0><<<gemm_grid, 256, 0, stream>>>(
        k, w_k, b_k, (void*)Kb, ROWS, D_MODEL, D_MODEL);
    gemm_bf16_wmma<float, 2><<<gemm_grid, 256, 0, stream>>>(
        v, w_v, b_v, (void*)Vt, ROWS, D_MODEL, D_MODEL);

    flash_attn_wmma<<<attn_grid, 256, 0, stream>>>(Qb, Kb, Vt, O);

    gemm_bf16_wmma<unsigned short, 3><<<gemm_grid, 256, 0, stream>>>(
        O, w_o, b_o, d_out, ROWS, D_MODEL, D_MODEL);
}
